// PatternFixLoss_9483287789852
// MI455X (gfx1250) — compile-verified, matrix operand order checked by reading, not run on hardware
//
#include <hip/hip_runtime.h>
#include <hip/hip_bf16.h>
#include <math.h>

// Problem constants (from reference)
#define B_    256
#define S_    512
#define V_    150
#define IN_   512
#define NPOS  (B_ * S_)          // 131072 positions
#define EOS_  2
#define EPSF  1e-8f

// Grid sizing
#define NB_MAIN 1024             // 1024 blocks * 8 waves = 8192 waves, 16 pos/wave
#define NB_PAT  (B_ * 3)         // 768: one block per (row, n)
#define NB_ENT  2048             // entropy streaming blocks

// Workspace layout.
// Floats: [0, WS_FLOATS)
#define WS_GATH   0
#define WS_MASK   (WS_GATH  + NB_MAIN)
#define WS_EOSLP  (WS_MASK  + NB_MAIN)
#define WS_EOSCT  (WS_EOSLP + NB_MAIN)
#define WS_PAT    (WS_EOSCT + NB_MAIN)
#define WS_ENT    (WS_PAT   + NB_PAT)
#define WS_FLOATS (WS_ENT   + NB_ENT)      // 6912 floats
// Ints (index into (int*)ws):
#define WI_HASEOS_T  WS_FLOATS             // 256 ints
#define WI_HASEOS_P  (WI_HASEOS_T + B_)    // 256 ints
#define WI_END       (WI_HASEOS_P + B_)
// Bytes: predicted tokens (V=150 fits u8)
#define TOK_BYTE_OFF ((size_t)WI_END * 4)  // 131072 bytes

typedef float v2f  __attribute__((ext_vector_type(2)));
typedef float v8f  __attribute__((ext_vector_type(8)));
typedef _Float16 v16h __attribute__((ext_vector_type(16)));

// ---------------------------------------------------------------------------
// Zero the atomicOr'd flag region each launch (harness does not re-poison).
__global__ void k_init(int* iws) {
  int i = blockIdx.x * blockDim.x + threadIdx.x;
  if (i < 2 * B_) iws[WI_HASEOS_T + i] = 0;
}

// ---------------------------------------------------------------------------
// One wave32 per (b,s) position: argmax over V=150, NLL gather, EOS softmax.
__global__ void k_main(const float* __restrict__ pred,
                       const int* __restrict__ tgt,
                       float* __restrict__ fws, int* __restrict__ iws,
                       unsigned char* __restrict__ tok_out) {
  const int lane = threadIdx.x & 31;
  const int wv   = threadIdx.x >> 5;                       // wave in block (0..7)
  const int wid  = blockIdx.x * (blockDim.x >> 5) + wv;    // 0..8191
  const int per_wave = NPOS / (NB_MAIN * 8);               // 16

  float acc_g = 0.f, acc_m = 0.f, acc_el = 0.f, acc_ec = 0.f;

  for (int q = 0; q < per_wave; ++q) {
    const int pos = wid * per_wave + q;
    const float* row = pred + (size_t)pos * V_;

    float vals[5];
    vals[0] = row[lane];
    float m = vals[0];
    int   mi = lane;
#pragma unroll
    for (int i = 1; i < 5; ++i) {
      int v = lane + 32 * i;
      vals[i] = (v < V_) ? row[v] : -INFINITY;
      if (vals[i] > m) { m = vals[i]; mi = v; }
    }
    // wave32 argmax butterfly (first-occurrence tie-break: smaller index wins)
#pragma unroll
    for (int off = 16; off >= 1; off >>= 1) {
      float om = __shfl_xor(m, off);
      int   oi = __shfl_xor(mi, off);
      if (om > m || (om == m && oi < mi)) { m = om; mi = oi; }
    }
    // sum of exp(x - max)
    float se = 0.f;
#pragma unroll
    for (int i = 0; i < 5; ++i) {
      int v = lane + 32 * i;
      if (v < V_) se += __expf(vals[i] - m);
    }
#pragma unroll
    for (int off = 16; off >= 1; off >>= 1) se += __shfl_xor(se, off);

    const float xe = __shfl(vals[0], EOS_);  // x[EOS] lives in lane 2, chunk 0
    const int tok = mi;

    if (lane == 0) {
      const int t = tgt[pos];
      if (t != 0) { acc_g += row[t]; acc_m += 1.0f; }   // NLL gather + mask
      if (t == EOS_) {
        float pe = __expf(xe - m) / se;                 // softmax prob of EOS
        acc_el += __logf(pe + EPSF);
        acc_ec += 1.0f;
        atomicOr(&iws[WI_HASEOS_T + pos / S_], 1);
      }
      tok_out[pos] = (unsigned char)tok;
      if (tok == EOS_) atomicOr(&iws[WI_HASEOS_P + pos / S_], 1);
    }
  }

  __shared__ float red[8][4];
  if (lane == 0) {
    red[wv][0] = acc_g; red[wv][1] = acc_m; red[wv][2] = acc_el; red[wv][3] = acc_ec;
  }
  __syncthreads();
  if (threadIdx.x == 0) {
    float a = 0.f, b = 0.f, c = 0.f, d = 0.f;
    for (int i = 0; i < 8; ++i) { a += red[i][0]; b += red[i][1]; c += red[i][2]; d += red[i][3]; }
    fws[WS_GATH  + blockIdx.x] = a;
    fws[WS_MASK  + blockIdx.x] = b;
    fws[WS_EOSLP + blockIdx.x] = c;
    fws[WS_EOSCT + blockIdx.x] = d;
  }
}

// ---------------------------------------------------------------------------
// One block per (row b, ngram n): count positions with >=2 earlier equal
// hashes (== sum over unique ngrams of max(count-2,0), same as the sort trick)
__global__ void k_pattern(const unsigned char* __restrict__ toks_g,
                          float* __restrict__ fws) {
  const int bn = blockIdx.x;
  const int b  = bn / 3;
  const int n  = 2 + (bn % 3);
  const int L  = S_ - n + 1;

  __shared__ int toks[S_];
  __shared__ int hs[S_];
  __shared__ int sred[256];

  const unsigned char* trow = toks_g + (size_t)b * S_;
  for (int i = threadIdx.x; i < S_; i += blockDim.x) toks[i] = (int)trow[i];
  __syncthreads();

  for (int i = threadIdx.x; i < L; i += blockDim.x) {
    int h = 0, p = 1;
    for (int k = 0; k < n; ++k) { h += toks[i + k] * p; p *= V_; }
    hs[i] = h;
  }
  __syncthreads();

  int rep = 0;
  for (int i = threadIdx.x; i < L; i += blockDim.x) {
    const int hi = hs[i];
    int cnt = 0;
    for (int j = 0; j < i; ++j) {
      cnt += (hs[j] == hi);
      if (cnt >= 2) break;
    }
    if (cnt >= 2) ++rep;
  }

  sred[threadIdx.x] = rep;
  __syncthreads();
  for (int s = 128; s > 0; s >>= 1) {
    if (threadIdx.x < s) sred[threadIdx.x] += sred[threadIdx.x + s];
    __syncthreads();
  }
  if (threadIdx.x == 0) fws[WS_PAT + bn] = (float)sred[0] * 0.1f * (float)n;
}

// ---------------------------------------------------------------------------
// Stream 268 MB of attention weights: sum of w*log(w+eps). Bandwidth-bound.
__global__ void k_ent(const float4* __restrict__ att, float* __restrict__ fws) {
  const int n4     = (B_ * S_ * IN_) / 4;      // 16,777,216
  const int stride = gridDim.x * blockDim.x;
  float s = 0.f;
  for (int i = blockIdx.x * blockDim.x + threadIdx.x; i < n4; i += stride) {
    __builtin_prefetch(att + i + 8 * stride, 0, 0);   // global_prefetch_b8
    float4 w = att[i];
    s += w.x * __logf(w.x + EPSF);
    s += w.y * __logf(w.y + EPSF);
    s += w.z * __logf(w.z + EPSF);
    s += w.w * __logf(w.w + EPSF);
  }
#pragma unroll
  for (int off = 16; off >= 1; off >>= 1) s += __shfl_xor(s, off);
  __shared__ float r[8];
  if ((threadIdx.x & 31) == 0) r[threadIdx.x >> 5] = s;
  __syncthreads();
  if (threadIdx.x == 0) {
    float t = 0.f;
    for (int i = 0; i < 8; ++i) t += r[i];
    fws[WS_ENT + blockIdx.x] = t;
  }
}

// ---------------------------------------------------------------------------
// Final combine. Entropy partials reduced with WMMA: D = ones x B + C makes
// every row of D accumulate column sums of B, so chained WMMAs reduce 64 (f32)
// or 512 (f16) values per instruction; finish with a width-16 shuffle.
__global__ void k_final(const float* __restrict__ fws,
                        const int* __restrict__ iws,
                        float* __restrict__ out) {
  __shared__ float s_ent;
  __shared__ float sr[256];
  const int lane = threadIdx.x & 31;

  if (threadIdx.x < 32) {   // wave 0 only; EXEC all-ones inside this wave
#if __has_builtin(__builtin_amdgcn_wmma_f32_16x16x4_f32)
    v2f a; a.x = 1.0f; a.y = 1.0f;                 // A = ones(16x4)
    v8f c = {};
#pragma unroll 4
    for (int ch = 0; ch < NB_ENT / 64; ++ch) {     // 32 chunks of 64
      v2f b;
      b.x = fws[WS_ENT + ch * 64 + lane];
      b.y = fws[WS_ENT + ch * 64 + 32 + lane];
      c = __builtin_amdgcn_wmma_f32_16x16x4_f32(false, a, false, b,
                                                (short)0, c, false, false);
    }
#else
    v16h a;
#pragma unroll
    for (int j = 0; j < 16; ++j) a[j] = (_Float16)1.0f;  // A = ones(16x32)
    v8f c = {};
    for (int ch = 0; ch < NB_ENT / 512; ++ch) {          // 4 chunks of 512
      v16h b;
#pragma unroll
      for (int j = 0; j < 16; ++j)
        b[j] = (_Float16)fws[WS_ENT + ch * 512 + j * 32 + lane];
      c = __builtin_amdgcn_wmma_f32_16x16x32_f16(false, a, false, b,
                                                 (short)0, c, false, false);
    }
#endif
    // D row 0 (lanes 0..15 of VGPR0) holds the 16 column sums
    float v = c[0];
#pragma unroll
    for (int off = 8; off >= 1; off >>= 1) v += __shfl_xor(v, off, 16);
    if (lane == 0) s_ent = v;
  }

  // Strided partial sums for the remaining accumulators
  float lg = 0.f, lm = 0.f, le = 0.f, lc = 0.f, lp = 0.f;
  for (int i = threadIdx.x; i < NB_MAIN; i += 256) {
    lg += fws[WS_GATH + i];  lm += fws[WS_MASK + i];
    le += fws[WS_EOSLP + i]; lc += fws[WS_EOSCT + i];
  }
  for (int i = threadIdx.x; i < NB_PAT; i += 256) lp += fws[WS_PAT + i];
  // 256 rows, 256 threads: no-EOS indicator
  const int r = threadIdx.x;
  float fl = (iws[WI_HASEOS_T + r] && !iws[WI_HASEOS_P + r]) ? 1.0f : 0.0f;

  auto blkred = [&](float v) -> float {
    sr[threadIdx.x] = v; __syncthreads();
    for (int s = 128; s > 0; s >>= 1) {
      if (threadIdx.x < s) sr[threadIdx.x] += sr[threadIdx.x + s];
      __syncthreads();
    }
    float res = sr[0]; __syncthreads();
    return res;
  };

  const float S_g  = blkred(lg);
  const float S_m  = blkred(lm);
  const float S_el = blkred(le);
  const float S_ec = blkred(lc);
  const float S_p  = blkred(lp);
  const float S_f  = blkred(fl);

  if (threadIdx.x == 0) {
    const float inv    = 1.0f / (float)(B_ * S_);
    const float main_l = -S_g / fmaxf(S_m, 1.0f);
    const float pattern = S_p / (float)B_;
    const float eos_loss = (S_ec > 0.0f) ? (-S_el * inv) : 0.0f;
    const float no_eos   = S_f / (float)B_;
    const float eos_total = eos_loss + 0.5f * no_eos;
    const float attention = 0.1f * s_ent * inv;  // s_ent = sum w*log(w+eps) < 0
    const float total = main_l + 0.1f * pattern + 0.2f * eos_total
                        + 0.05f * attention;
    out[0] = total; out[1] = main_l; out[2] = pattern;
    out[3] = eos_total; out[4] = attention;
  }
}

// ---------------------------------------------------------------------------
extern "C" void kernel_launch(void* const* d_in, const int* in_sizes, int n_in,
                              void* d_out, int out_size, void* d_ws, size_t ws_size,
                              hipStream_t stream) {
  const float* pred = (const float*)d_in[0];
  const int*   tgt  = (const int*)d_in[1];
  const float* att  = (const float*)d_in[2];

  float* fws = (float*)d_ws;
  int*   iws = (int*)d_ws;
  unsigned char* tok = (unsigned char*)d_ws + TOK_BYTE_OFF;
  float* out = (float*)d_out;

  k_init   <<<2,       256, 0, stream>>>(iws);
  k_main   <<<NB_MAIN, 256, 0, stream>>>(pred, tgt, fws, iws, tok);
  k_pattern<<<NB_PAT,  256, 0, stream>>>(tok, fws);
  k_ent    <<<NB_ENT,  256, 0, stream>>>((const float4*)att, fws);
  k_final  <<<1,       256, 0, stream>>>(fws, iws, out);
}
